// SimpleEncoder_14293651161200
// MI455X (gfx1250) — matrix operand order repaired and checked by reading, not compiled, not run on hardware
//
#include <hip/hip_runtime.h>

typedef __attribute__((ext_vector_type(16))) _Float16 v16h;
typedef __attribute__((ext_vector_type(8)))  float    v8f;

#define BB   16
#define NPTS 8192
#define NG   512
#define KNN  32

// ---------------------------------------------------------------------------
// WMMA fragment loaders (gfx1250 wave32; contiguous k = 16*(lane/16) + s in A and B)
// ---------------------------------------------------------------------------

// A: 16x32 fp16 tile from row-major fp16 LDS (stride in elements).
// lanes 0-15 = rows, K {0..15}; lanes 16-31 = rows, K {16..31}
__device__ __forceinline__ v16h load_a_lds(const _Float16* sm, int stride,
                                           int rowbase, int k0) {
  const int lane = threadIdx.x & 31;
  const int row  = rowbase + (lane & 15);
  const int kb   = (lane < 16) ? 0 : 16;
  const _Float16* p = sm + row * stride + k0;
  v16h a;
#pragma unroll
  for (int q = 0; q < 8; ++q) {
    int k = kb + 2 * q;
    a[2 * q]     = p[k];
    a[2 * q + 1] = p[k + 1];
  }
  return a;
}

// B: 32x16 fp16 tile (B = W^T) from row-major fp16 weights [outC][inC].
// lane = output column (lanes 0-15: K=k0..k0+15, lanes 16-31: K=k0+16..k0+31)
__device__ __forceinline__ v16h load_b_w(const _Float16* w, int inC,
                                         int n0, int k0) {
  const int lane = threadIdx.x & 31;
  const int n = n0 + (lane & 15);
  const int k = k0 + ((lane < 16) ? 0 : 16);
  const _Float16* p = w + (size_t)n * inC + k;   // 16 contiguous halfs
  v16h b;
#pragma unroll
  for (int j = 0; j < 16; ++j) b[j] = p[j];
  return b;
}

// Row-block GEMM: this wave owns M-tile `mt`; keeps NT accumulators live and
// loads each A fragment once per k-step (A reuse across all N tiles).
template <int NT>
__device__ __forceinline__ void gemm_rowblock(const _Float16* smA, int astride,
                                              int mt, const _Float16* w, int inC,
                                              int ksteps, int ntBase, int ntStep,
                                              v8f* acc) {
  for (int ks = 0; ks < ksteps; ++ks) {
    v16h a = load_a_lds(smA, astride, mt * 16, ks * 32);
#pragma unroll
    for (int j = 0; j < NT; ++j) {
      v16h b = load_b_w(w, inC, (ntBase + j * ntStep) * 16, ks * 32);
      acc[j] = __builtin_amdgcn_wmma_f32_16x16x32_f16(false, a, false, b,
                                                      (short)0, acc[j],
                                                      false, false);
    }
  }
}

// Wave32 argmax reduce (max dist, lowest index wins ties)
__device__ __forceinline__ void wave_argmax(float& d, int& i) {
#pragma unroll
  for (int off = 16; off > 0; off >>= 1) {
    float od = __shfl_down(d, off, 32);
    int   oi = __shfl_down(i, off, 32);
    if (od > d || (od == d && oi < i)) { d = od; i = oi; }
  }
}

// Wave32 argmin reduce (min dist, lowest index wins ties)
__device__ __forceinline__ void wave_argmin(float& d, int& i) {
#pragma unroll
  for (int off = 16; off > 0; off >>= 1) {
    float od = __shfl_down(d, off, 32);
    int   oi = __shfl_down(i, off, 32);
    if (od < d || (od == d && oi < i)) { d = od; i = oi; }
  }
}

// ---------------------------------------------------------------------------
// 1) Farthest point sampling: one block per batch, points in registers,
//    shuffle-based reduction (2 barriers per iteration instead of ~11)
// ---------------------------------------------------------------------------
__global__ __launch_bounds__(1024)
void fps_kernel(const float* __restrict__ xyz, float* __restrict__ center) {
  const int b    = blockIdx.x;
  const int tid  = threadIdx.x;
  const int lane = tid & 31;
  const int wave = tid >> 5;
  const float* px = xyz + (size_t)b * NPTS * 3;

  float X[8], Y[8], Z[8], D[8];
#pragma unroll
  for (int j = 0; j < 8; ++j) {
    int i = j * 1024 + tid;
    X[j] = px[i * 3 + 0]; Y[j] = px[i * 3 + 1]; Z[j] = px[i * 3 + 2];
    D[j] = 1e10f;
  }

  __shared__ float cx, cy, cz;
  __shared__ float sd[32];
  __shared__ int   si[32];

  if (tid == 0) {
    cx = px[0]; cy = px[1]; cz = px[2];
    float* c0 = center + (size_t)b * NG * 3;
    c0[0] = cx; c0[1] = cy; c0[2] = cz;
  }
  __syncthreads();

  for (int it = 1; it < NG; ++it) {
    const float bx = cx, by = cy, bz = cz;
    float bestd = -1.0f; int besti = 0;
#pragma unroll
    for (int j = 0; j < 8; ++j) {
      float dx = X[j] - bx, dy = Y[j] - by, dz = Z[j] - bz;
      float d = dx * dx + dy * dy + dz * dz;
      D[j] = fminf(D[j], d);
      int i = j * 1024 + tid;
      if (D[j] > bestd || (D[j] == bestd && i < besti)) { bestd = D[j]; besti = i; }
    }
    wave_argmax(bestd, besti);
    if (lane == 0) { sd[wave] = bestd; si[wave] = besti; }
    __syncthreads();
    if (tid < 32) {
      float d = sd[tid]; int i = si[tid];
      wave_argmax(d, i);
      if (tid == 0) {
        cx = px[i * 3 + 0]; cy = px[i * 3 + 1]; cz = px[i * 3 + 2];
        float* c = center + ((size_t)b * NG + it) * 3;
        c[0] = cx; c[1] = cy; c[2] = cz;
      }
    }
    __syncthreads();
  }
}

// ---------------------------------------------------------------------------
// 2) kNN (k=32): one block per center, distances staged in LDS,
//    shuffle-based min-reduction
// ---------------------------------------------------------------------------
__global__ __launch_bounds__(256)
void knn_kernel(const float* __restrict__ xyz, const float* __restrict__ center,
                int* __restrict__ idxout) {
  const int grp  = blockIdx.x;          // b*NG + g
  const int b    = grp >> 9;
  const int tid  = threadIdx.x;
  const int lane = tid & 31;
  const int wave = tid >> 5;
  const float* px = xyz + (size_t)b * NPTS * 3;

  __shared__ float dist[NPTS];
  __shared__ float rd[8];
  __shared__ int   ri[8];

  const float cx = center[grp * 3 + 0];
  const float cy = center[grp * 3 + 1];
  const float cz = center[grp * 3 + 2];

  for (int j = 0; j < 32; ++j) {
    int i = j * 256 + tid;
    float dx = px[i * 3 + 0] - cx, dy = px[i * 3 + 1] - cy, dz = px[i * 3 + 2] - cz;
    dist[i] = dx * dx + dy * dy + dz * dz;
  }
  __syncthreads();

  for (int kk = 0; kk < KNN; ++kk) {
    float bd = 3.0e38f; int bi = 0;
    for (int j = 0; j < 32; ++j) {
      int i = j * 256 + tid;
      float d = dist[i];
      if (d < bd || (d == bd && i < bi)) { bd = d; bi = i; }
    }
    wave_argmin(bd, bi);
    if (lane == 0) { rd[wave] = bd; ri[wave] = bi; }
    __syncthreads();
    if (tid == 0) {
      float d = rd[0]; int w = ri[0];
#pragma unroll
      for (int j = 1; j < 8; ++j) {
        if (rd[j] < d || (rd[j] == d && ri[j] < w)) { d = rd[j]; w = ri[j]; }
      }
      idxout[(size_t)grp * KNN + kk] = w;
      dist[w] = 3.4e38f;
    }
    __syncthreads();
  }
}

// ---------------------------------------------------------------------------
// 3) Prep: zero BN accumulators, fp32 weights -> fp16 (L2-resident)
// ---------------------------------------------------------------------------
__global__ void prep_kernel(const float* __restrict__ w2, const float* __restrict__ w3,
                            const float* __restrict__ w4,
                            _Float16* __restrict__ w2h, _Float16* __restrict__ w3h,
                            _Float16* __restrict__ w4h, float* __restrict__ stats) {
  int i = blockIdx.x * 256 + threadIdx.x;
  if (i < 2560)      stats[i] = 0.0f;
  if (i < 256 * 128) w2h[i] = (_Float16)w2[i];
  if (i < 512 * 512) w3h[i] = (_Float16)w3[i];
  if (i < 384 * 512) w4h[i] = (_Float16)w4[i];
}

// ---------------------------------------------------------------------------
// 4) Gather + pointwise 3->128 + BN1 partial sums
// ---------------------------------------------------------------------------
__global__ __launch_bounds__(128)
void enc1_kernel(const float* __restrict__ xyz, const int* __restrict__ idx,
                 const float* __restrict__ w1, const float* __restrict__ b1,
                 float* __restrict__ x1, float* __restrict__ sum1,
                 float* __restrict__ sq1) {
  const int grp = blockIdx.x;
  const int b   = grp >> 9;
  const int tid = threadIdx.x;   // == channel, 128 threads

  __shared__ float nb[KNN][3];
  if (tid < KNN) {
    int i = idx[(size_t)grp * KNN + tid];
    const float* p = xyz + ((size_t)b * NPTS + i) * 3;
    nb[tid][0] = p[0]; nb[tid][1] = p[1]; nb[tid][2] = p[2];
  }
  __syncthreads();

  const float wa = w1[tid * 3 + 0], wb = w1[tid * 3 + 1], wc = w1[tid * 3 + 2];
  const float bc = b1[tid];
  float s = 0.0f, q = 0.0f;
  float* xo = x1 + (size_t)grp * KNN * 128;
  for (int r = 0; r < KNN; ++r) {
    float v = nb[r][0] * wa + nb[r][1] * wb + nb[r][2] * wc + bc;
    xo[r * 128 + tid] = v;
    s += v; q += v * v;
  }
  atomicAdd(&sum1[tid], s);
  atomicAdd(&sq1[tid], q);
}

// ---------------------------------------------------------------------------
// 5) Fold BN mean/var into per-channel scale/shift
// ---------------------------------------------------------------------------
__global__ void bnfin_kernel(const float* __restrict__ sum, const float* __restrict__ sq,
                             const float* __restrict__ gamma, const float* __restrict__ beta,
                             float* __restrict__ scale, float* __restrict__ shift, int C) {
  int c = blockIdx.x * blockDim.x + threadIdx.x;
  if (c >= C) return;
  const float inv = 1.0f / (float)(BB * NG * KNN);
  float m = sum[c] * inv;
  float v = sq[c] * inv - m * m;
  float a = gamma[c] * rsqrtf(v + 1e-5f);
  scale[c] = a;
  shift[c] = beta[c] - m * a;
}

// ---------------------------------------------------------------------------
// 6) Core WMMA kernel: BN1+ReLU -> mm2 -> max/concat (LDS) -> mm3 -> x3 + BN2 sums
//    One block (8 waves) per group; waves 0-3 own rows 0-15, waves 4-7 rows 16-31.
// ---------------------------------------------------------------------------
__global__ __launch_bounds__(256)
void enc3_kernel(const float* __restrict__ x1, const float* __restrict__ ab1,
                 const _Float16* __restrict__ w2h, const float* __restrict__ b2,
                 const _Float16* __restrict__ w3h, const float* __restrict__ b3,
                 _Float16* __restrict__ x3h, float* __restrict__ sum2,
                 float* __restrict__ sq2) {
  const int grp  = blockIdx.x;
  const int tid  = threadIdx.x;
  const int wave = tid >> 5;
  const int lane = tid & 31;
  const int mt   = wave >> 2;   // M-tile owned by this wave
  const int wq   = wave & 3;    // N-tile base within quarter

  __shared__ _Float16 smA[KNN * 128];   // BN1+ReLU(x1), fp16      (8 KB)
  __shared__ _Float16 smC[KNN * 512];   // concat [fg | f2], fp16  (32 KB)

  // Pull weight streams toward the caches while we stage A
  __builtin_prefetch(w2h + (size_t)tid * 128, 0, 1);
  __builtin_prefetch(w3h + (size_t)tid * 1024, 0, 1);

  // BN1 + ReLU -> fp16 A-operand
  {
    const float* xi = x1 + (size_t)grp * KNN * 128;
    for (int i = tid; i < KNN * 128; i += 256) {
      int c = i & 127;
      float v = xi[i] * ab1[c] + ab1[128 + c];
      smA[i] = (_Float16)fmaxf(v, 0.0f);
    }
  }
  __syncthreads();

  // mm2: [32x128] @ [128x256] -> f2 in smC[:, 256:512]; 4 N-tiles per wave
  {
    v8f acc[4] = {};
    gemm_rowblock<4>(smA, 128, mt, w2h, 128, 4, wq, 4, acc);
    const int rb = mt * 16 + ((lane < 16) ? 0 : 8);
#pragma unroll
    for (int j = 0; j < 4; ++j) {
      const int col = (wq + 4 * j) * 16 + (lane & 15);
      const float bias = b2[col];
#pragma unroll
      for (int i = 0; i < 8; ++i)
        smC[(rb + i) * 512 + 256 + col] = (_Float16)(acc[j][i] + bias);
    }
  }
  __syncthreads();

  // fg = max over K rows, broadcast into smC[:, 0:256]
  {
    float m = -3.4e38f;
    for (int r = 0; r < KNN; ++r) m = fmaxf(m, (float)smC[r * 512 + 256 + tid]);
    _Float16 mh = (_Float16)m;
    for (int r = 0; r < KNN; ++r) smC[r * 512 + tid] = mh;
  }
  __syncthreads();

  // mm3: [32x512] @ [512x512] -> x3 (pre-BN2) fp16 to HBM + BN2 sums
  // 8 N-tiles per wave, one A-fragment load per k-step shared by all 8
  {
    v8f acc[8] = {};
    gemm_rowblock<8>(smC, 512, mt, w3h, 512, 16, wq, 4, acc);
    _Float16* xo = x3h + (size_t)grp * KNN * 512;
    const int rb = mt * 16 + ((lane < 16) ? 0 : 8);
#pragma unroll
    for (int j = 0; j < 8; ++j) {
      const int col = (wq + 4 * j) * 16 + (lane & 15);
      const float bias = b3[col];
      float s = 0.0f, q = 0.0f;
#pragma unroll
      for (int i = 0; i < 8; ++i) {
        float v = acc[j][i] + bias;
        xo[(rb + i) * 512 + col] = (_Float16)v;
        s += v; q += v * v;
      }
      s += __shfl_xor(s, 16, 32);
      q += __shfl_xor(q, 16, 32);
      if (lane < 16) {
        atomicAdd(&sum2[col], s);
        atomicAdd(&sq2[col], q);
      }
    }
  }
}

// ---------------------------------------------------------------------------
// 7) BN2+ReLU -> mm4 (WMMA) -> max over K -> + b4 -> feats
// ---------------------------------------------------------------------------
__global__ __launch_bounds__(256)
void enc5_kernel(const _Float16* __restrict__ x3h, const float* __restrict__ ab2,
                 const _Float16* __restrict__ w4h, const float* __restrict__ b4,
                 float* __restrict__ feats) {
  const int grp  = blockIdx.x;
  const int tid  = threadIdx.x;
  const int wave = tid >> 5;
  const int lane = tid & 31;
  const int mt   = wave >> 2;
  const int wq   = wave & 3;

  __shared__ _Float16 smA[KNN * 512];   // BN2+ReLU(x3) fp16 (32 KB)
  __shared__ float pmax[2 * 384];       // per-(mtile, channel) partial max

  __builtin_prefetch(w4h + (size_t)tid * 768, 0, 1);

  const _Float16* xi = x3h + (size_t)grp * KNN * 512;
  for (int i = tid; i < KNN * 512; i += 256) {
    int c = i & 511;
    float v = (float)xi[i] * ab2[c] + ab2[512 + c];
    smA[i] = (_Float16)fmaxf(v, 0.0f);
  }
  __syncthreads();

  // mm4: [32x512] @ [512x384]; 6 N-tiles per wave, shared A fragments
  {
    v8f acc[6] = {};
    gemm_rowblock<6>(smA, 512, mt, w4h, 512, 16, wq, 4, acc);
#pragma unroll
    for (int j = 0; j < 6; ++j) {
      const int nt = wq + 4 * j;
      float m = -3.4e38f;
#pragma unroll
      for (int i = 0; i < 8; ++i) m = fmaxf(m, acc[j][i]);
      m = fmaxf(m, __shfl_xor(m, 16, 32));   // combine rows 0-7 with 8-15
      if (lane < 16) pmax[mt * 384 + nt * 16 + lane] = m;
    }
  }
  __syncthreads();

  for (int c = tid; c < 384; c += 256)
    feats[(size_t)grp * 384 + c] = fmaxf(pmax[c], pmax[384 + c]) + b4[c];
}

// ---------------------------------------------------------------------------
// Host launch
// ---------------------------------------------------------------------------
extern "C" void kernel_launch(void* const* d_in, const int* in_sizes, int n_in,
                              void* d_out, int out_size, void* d_ws, size_t ws_size,
                              hipStream_t stream) {
  (void)in_sizes; (void)n_in; (void)out_size; (void)ws_size;

  const float* xyz = (const float*)d_in[0];
  const float* w1  = (const float*)d_in[2];
  const float* b1  = (const float*)d_in[3];
  const float* g1  = (const float*)d_in[4];
  const float* be1 = (const float*)d_in[5];
  const float* w2  = (const float*)d_in[6];
  const float* b2  = (const float*)d_in[7];
  const float* w3  = (const float*)d_in[8];
  const float* b3  = (const float*)d_in[9];
  const float* g2  = (const float*)d_in[10];
  const float* be2 = (const float*)d_in[11];
  const float* w4  = (const float*)d_in[12];
  const float* b4  = (const float*)d_in[13];

  float* center = (float*)d_out;                // [16,512,3]
  float* feats  = center + BB * NG * 3;         // [16,512,384]

  // workspace layout (floats)
  float* wsf  = (float*)d_ws;
  float* sum1 = wsf;            // 128
  float* sq1  = wsf + 128;      // 128
  float* ab1  = wsf + 256;      // scale128 | shift128
  float* sum2 = wsf + 512;      // 512
  float* sq2  = wsf + 1024;     // 512
  float* ab2  = wsf + 1536;     // scale512 | shift512  (ends at 2560)
  int*   idxb = (int*)(wsf + 2560);                       // 262144 ints
  _Float16* w2h = (_Float16*)(wsf + 2560 + 262144);       // 32768 halfs
  _Float16* w3h = w2h + 256 * 128;                        // 262144 halfs
  _Float16* w4h = w3h + 512 * 512;                        // 196608 halfs
  float*    x1  = (float*)(w4h + 384 * 512);              // 262144*128 f32
  _Float16* x3h = (_Float16*)(x1 + (size_t)BB * NG * KNN * 128); // 262144*512 f16

  fps_kernel  <<<BB,        1024, 0, stream>>>(xyz, center);
  knn_kernel  <<<BB * NG,   256,  0, stream>>>(xyz, center, idxb);
  prep_kernel <<<1024,      256,  0, stream>>>(w2, w3, w4, w2h, w3h, w4h, wsf);
  enc1_kernel <<<BB * NG,   128,  0, stream>>>(xyz, idxb, w1, b1, x1, sum1, sq1);
  bnfin_kernel<<<1,         128,  0, stream>>>(sum1, sq1, g1, be1, ab1, ab1 + 128, 128);
  enc3_kernel <<<BB * NG,   256,  0, stream>>>(x1, ab1, w2h, b2, w3h, b3, x3h, sum2, sq2);
  bnfin_kernel<<<2,         256,  0, stream>>>(sum2, sq2, g2, be2, ab2, ab2 + 512, 512);
  enc5_kernel <<<BB * NG,   256,  0, stream>>>(x3h, ab2, w4h, b4, feats);
}
